// VectorOut_13185549598888
// MI455X (gfx1250) — matrix operand order repaired and checked by reading, not compiled, not run on hardware
//
#include <hip/hip_runtime.h>

typedef __attribute__((ext_vector_type(16))) _Float16 v16h;
typedef __attribute__((ext_vector_type(8)))  _Float16 v8h;
typedef __attribute__((ext_vector_type(8)))  float    v8f;

#define N_NODES      500000
#define NODE_DIM     128
#define HID          64
#define L1_IN        64
#define L1_HID       32
#define NUM_GRAPHS   4096
#define WAVES_PB     4
#define NODES_PW     16
#define NODES_PB     (WAVES_PB * NODES_PW)

__global__ void zero_out_kernel(float* __restrict__ out, int n) {
    int i = blockIdx.x * blockDim.x + threadIdx.x;
    if (i < n) out[i] = 0.0f;
}

__global__ void __launch_bounds__(128)
fused_equiv_readout(const float* __restrict__ xs,     // [N,128]
                    const float* __restrict__ xsph,   // [N,480]
                    const int*   __restrict__ bidx,   // [N]
                    const float* __restrict__ W1,     // [128,64]
                    const float* __restrict__ b1,     // [64]
                    const float* __restrict__ W2,     // [64,1]
                    const float* __restrict__ b2,     // [1]
                    const float* __restrict__ Ws1,    // [64,32]
                    const float* __restrict__ Ws2,    // [32,1]
                    float*       __restrict__ out)    // [4096,3]
{
    // LDS: W1^T (64x128 f16) = 16KB, Ws1^T (32x64 f16) = 4KB,
    // per-wave deinterleaved spherical tile 16x3x64 f16 = 6KB * 4 = 24KB,
    // per-wave scalar[16] f32.  Total ~44.3KB.
    __shared__ __align__(32) _Float16 w1t[HID * NODE_DIM];        // [n][k]
    __shared__ __align__(32) _Float16 wst[L1_HID * L1_IN];        // [o][c]
    __shared__ __align__(32) _Float16 sph[WAVES_PB][NODES_PW * 3 * L1_IN]; // [nd][i][c]
    __shared__ float sclr[WAVES_PB][NODES_PW];

    const int tid  = threadIdx.x;
    const int wave = tid >> 5;
    const int lane = tid & 31;
    const int nl   = lane & 15;   // N-column (B/C/D) or M-row (A)
    const int h    = lane >> 4;   // lane-half -> K-half selector

    // ---- Stage weights as f16, transposed so B-fragments are contiguous ----
    for (int idx = tid; idx < NODE_DIM * HID; idx += blockDim.x) {
        int k = idx >> 6, n = idx & 63;               // W1 row-major [k][n]
        w1t[n * NODE_DIM + k] = (_Float16)W1[idx];
    }
    for (int idx = tid; idx < L1_IN * L1_HID; idx += blockDim.x) {
        int c = idx >> 5, o = idx & 31;               // Ws1 row-major [c][o]
        wst[o * L1_IN + c] = (_Float16)Ws1[idx];
    }

    // ---- Stage & deinterleave spherical slice: [node][3c+i] -> [node][i][c] ----
    const long nodeBase = (long)blockIdx.x * NODES_PB + (long)wave * NODES_PW;
    #pragma unroll
    for (int q = 0; q < 96; q += 4) {
        int flat = lane * 96 + q;                     // 3072 halves per wave
        int nd   = flat / 192;
        int rem  = flat - nd * 192;
        long gn  = nodeBase + nd; if (gn > N_NODES - 1) gn = N_NODES - 1;
        const float4 f = *(const float4*)(xsph + gn * 480 + 128 + rem);
        float vv[4] = {f.x, f.y, f.z, f.w};
        #pragma unroll
        for (int j = 0; j < 4; ++j) {
            int rr = rem + j;
            int c  = rr / 3;
            int ii = rr - 3 * c;
            sph[wave][nd * 192 + ii * 64 + c] = (_Float16)vv[j];
        }
    }
    __syncthreads();

    // ================= Scalar MLP path: S = X(16x128) @ W1(128x64) ==========
    long nodeA = nodeBase + nl; if (nodeA > N_NODES - 1) nodeA = N_NODES - 1;
    const float* xrow = xs + nodeA * NODE_DIM;

    v8f accS[4];
    #pragma unroll
    for (int t = 0; t < 4; ++t) accS[t] = (v8f){};

    #pragma unroll
    for (int kt = 0; kt < 4; ++kt) {
        // A fragment: lane holds M=nl; K in {k0..k0+7} U {k0+16..k0+23}, k0 = kt*32+h*8
        const int k0 = kt * 32 + h * 8;
        float4 f0 = *(const float4*)(xrow + k0);
        float4 f1 = *(const float4*)(xrow + k0 + 4);
        float4 f2 = *(const float4*)(xrow + k0 + 16);
        float4 f3 = *(const float4*)(xrow + k0 + 20);
        v16h a;
        a[0]=(_Float16)f0.x; a[1]=(_Float16)f0.y; a[2]=(_Float16)f0.z; a[3]=(_Float16)f0.w;
        a[4]=(_Float16)f1.x; a[5]=(_Float16)f1.y; a[6]=(_Float16)f1.z; a[7]=(_Float16)f1.w;
        a[8]=(_Float16)f2.x; a[9]=(_Float16)f2.y; a[10]=(_Float16)f2.z; a[11]=(_Float16)f2.w;
        a[12]=(_Float16)f3.x; a[13]=(_Float16)f3.y; a[14]=(_Float16)f3.z; a[15]=(_Float16)f3.w;
        #pragma unroll
        for (int t = 0; t < 4; ++t) {
            // B fragment: K in {k0..k0+7} U {k0+16..k0+23}, k0 = kt*32+h*8, from w1t[col][k]
            const _Float16* bp = &w1t[(t * 16 + nl) * NODE_DIM + kt * 32 + h * 8];
            v8h blo = *(const v8h*)(bp);
            v8h bhi = *(const v8h*)(bp + 16);
            v16h b;
            for (int e = 0; e < 8; ++e) { b[e] = blo[e]; b[8 + e] = bhi[e]; }
            accS[t] = __builtin_amdgcn_wmma_f32_16x16x32_f16(
                false, a, false, b, (short)0, accS[t], false, false);
        }
    }

    // bias + SiLU + dot with W2, reduce over N (across 16 lanes in each half)
    float p[8];
    #pragma unroll
    for (int r = 0; r < 8; ++r) p[r] = 0.0f;
    #pragma unroll
    for (int t = 0; t < 4; ++t) {
        int n = t * 16 + nl;
        float bb = b1[n], w2 = W2[n];
        #pragma unroll
        for (int r = 0; r < 8; ++r) {
            float s  = accS[t][r] + bb;
            float si = s / (1.0f + __expf(-s));       // silu
            p[r] += si * w2;
        }
    }
    #pragma unroll
    for (int r = 0; r < 8; ++r) {
        p[r] += __shfl_xor(p[r], 1);
        p[r] += __shfl_xor(p[r], 2);
        p[r] += __shfl_xor(p[r], 4);
        p[r] += __shfl_xor(p[r], 8);
    }
    float b2v = b2[0];
    if (nl == 0) {
        #pragma unroll
        for (int r = 0; r < 8; ++r) sclr[wave][h * 8 + r] = p[r] + b2v;
    }
    __syncthreads();

    // ============ Spherical path: H_i = X1e_i(16x64) @ Ws1(64x32), i=0..2 ====
    v8f accH[3][2];
    #pragma unroll
    for (int i = 0; i < 3; ++i)
        #pragma unroll
        for (int t = 0; t < 2; ++t) accH[i][t] = (v8f){};

    const _Float16* mysph = &sph[wave][nl * 192];
    #pragma unroll
    for (int kt = 0; kt < 2; ++kt) {
        v16h a3[3];
        #pragma unroll
        for (int i = 0; i < 3; ++i) {
            const _Float16* base = mysph + i * 64 + kt * 32 + h * 8;
            v8h lo = *(const v8h*)(base);
            v8h hi = *(const v8h*)(base + 16);
            v16h a;
            #pragma unroll
            for (int e = 0; e < 8; ++e) { a[e] = lo[e]; a[8 + e] = hi[e]; }
            a3[i] = a;
        }
        #pragma unroll
        for (int t = 0; t < 2; ++t) {
            const _Float16* bp = &wst[(t * 16 + nl) * L1_IN + kt * 32 + h * 8];
            v8h blo = *(const v8h*)(bp);
            v8h bhi = *(const v8h*)(bp + 16);
            v16h b;
            for (int e = 0; e < 8; ++e) { b[e] = blo[e]; b[8 + e] = bhi[e]; }
            #pragma unroll
            for (int i = 0; i < 3; ++i) {
                accH[i][t] = __builtin_amdgcn_wmma_f32_16x16x32_f16(
                    false, a3[i], false, b, (short)0, accH[i][t], false, false);
            }
        }
    }

    // gate = sigmoid(||h||), h scaled by 1/sqrt(64); contract over o with Ws2
    float v[3][8];
    #pragma unroll
    for (int i = 0; i < 3; ++i)
        #pragma unroll
        for (int r = 0; r < 8; ++r) v[i][r] = 0.0f;

    #pragma unroll
    for (int t = 0; t < 2; ++t) {
        int o = t * 16 + nl;
        float w = Ws2[o];
        #pragma unroll
        for (int r = 0; r < 8; ++r) {
            float h0 = accH[0][t][r] * 0.125f;
            float h1 = accH[1][t][r] * 0.125f;
            float h2 = accH[2][t][r] * 0.125f;
            float nrm = sqrtf(h0 * h0 + h1 * h1 + h2 * h2);
            float g = 1.0f / (1.0f + __expf(-nrm));
            float gw = g * w;
            v[0][r] += h0 * gw;
            v[1][r] += h1 * gw;
            v[2][r] += h2 * gw;
        }
    }
    #pragma unroll
    for (int i = 0; i < 3; ++i)
        #pragma unroll
        for (int r = 0; r < 8; ++r) {
            v[i][r] += __shfl_xor(v[i][r], 1);
            v[i][r] += __shfl_xor(v[i][r], 2);
            v[i][r] += __shfl_xor(v[i][r], 4);
            v[i][r] += __shfl_xor(v[i][r], 8);
        }

    // ============== atom_out = vec[[2,0,1]] * scalar; segment_sum ============
    const float inv32 = 0.17677669529663687f;  // 1/sqrt(32)
    if (nl == 0) {                              // lanes 0 and 16 write 8 nodes each
        #pragma unroll
        for (int r = 0; r < 8; ++r) {
            int m = h * 8 + r;
            long gn = nodeBase + m;
            if (gn < N_NODES) {
                int g = bidx[gn];
                float s = sclr[wave][m] * inv32;
                atomicAdd(&out[g * 3 + 0], v[2][r] * s);
                atomicAdd(&out[g * 3 + 1], v[0][r] * s);
                atomicAdd(&out[g * 3 + 2], v[1][r] * s);
            }
        }
    }
}

extern "C" void kernel_launch(void* const* d_in, const int* in_sizes, int n_in,
                              void* d_out, int out_size, void* d_ws, size_t ws_size,
                              hipStream_t stream) {
    const float* xs   = (const float*)d_in[0];   // x_scalar   [N,128]
    const float* xsph = (const float*)d_in[1];   // x_spherical[N,480]
    // d_in[2] = coord (unused by reference output)
    const int*   bidx = (const int*)d_in[3];     // batch_index[N]
    // d_in[4] = num_segments (constant 4096)
    const float* W1   = (const float*)d_in[5];
    const float* b1   = (const float*)d_in[6];
    const float* W2   = (const float*)d_in[7];
    const float* b2   = (const float*)d_in[8];
    const float* Ws1  = (const float*)d_in[9];
    const float* Ws2  = (const float*)d_in[10];
    float* out = (float*)d_out;

    const int out_elems = NUM_GRAPHS * 3;
    hipLaunchKernelGGL(zero_out_kernel, dim3((out_elems + 255) / 256), dim3(256),
                       0, stream, out, out_elems);

    const int blocks = (N_NODES + NODES_PB - 1) / NODES_PB;
    hipLaunchKernelGGL(fused_equiv_readout, dim3(blocks), dim3(128), 0, stream,
                       xs, xsph, bidx, W1, b1, W2, b2, Ws1, Ws2, out);
}
